// RNNDecoder_67456756351173
// MI455X (gfx1250) — matrix operand order repaired and checked by reading, not compiled, not run on hardware
//
#include <hip/hip_runtime.h>
#include <hip/hip_bf16.h>
#include <math.h>

// ---------------------------------------------------------------------------
// Types for CDNA5 WMMA (wave32)
// ---------------------------------------------------------------------------
typedef __bf16 bhalf;
typedef __attribute__((ext_vector_type(16))) __bf16 v16bf;
typedef __attribute__((ext_vector_type(8)))  float  v8f;

namespace {
constexpr int kV = 32000, kE = 512, kH = 1024, kA = 1024;
constexpr int kB = 16, kT = 64, kS = 64;
constexpr int kG = 3 * kH;        // 3072 (gate width)
constexpr int kKin0 = kE + kH;    // 1536 (GRU0 input width: [x | ctx])
}

// ---------------------------------------------------------------------------
// WMMA fragment loaders (per ISA 7.12.2 layouts, wave32)
// A 16x32 bf16: lane(0-15)=M, half=lane>>4; per-lane K = {8h..8h+7} U {16+8h..16+8h+7}
// B 32x16 bf16: lane&15=N;    per-lane K = {8h..8h+7} U {16+8h..16+8h+7}
// ---------------------------------------------------------------------------
__device__ __forceinline__ v16bf load_fragA(const bhalf* p) {
  union { v16bf v; uint4 q[2]; } u;
  u.q[0] = *reinterpret_cast<const uint4*>(p);        // K = base .. base+7
  u.q[1] = *reinterpret_cast<const uint4*>(p + 16);   // K = base+16 .. base+23
  return u.v;
}
__device__ __forceinline__ v16bf load_fragB(const bhalf* p) {
  union { v16bf v; uint4 q[2]; } u;
  u.q[0] = *reinterpret_cast<const uint4*>(p);        // K = base .. base+7
  u.q[1] = *reinterpret_cast<const uint4*>(p + 16);   // K = base+16 .. base+23
  return u.v;
}

// ---------------------------------------------------------------------------
// Generic bf16 WMMA GEMM:  out[M,N] (f32) = A[M,K] (bf16) * W[N,K]^T (bf16)
// One wave owns a 16x64 output strip (4 N-tiles). M%16==0, N%64==0, K%32==0.
// ---------------------------------------------------------------------------
__global__ void k_gemm_bf16(const bhalf* __restrict__ A, int ldA,
                            const bhalf* __restrict__ W, int ldW,
                            float* __restrict__ out, int ldO,
                            int M, int N, int K)
{
  const int lane = threadIdx.x & 31;
  const int wid  = (int)((blockIdx.x * blockDim.x + threadIdx.x) >> 5);
  const int nGrp = N >> 6;
  if (wid >= (M >> 4) * nGrp) return;
  const int tm   = wid / nGrp;
  const int ng   = wid - tm * nGrp;
  const int half = lane >> 4;
  const int l15  = lane & 15;

  const bhalf* aP = A + (size_t)(tm * 16 + l15) * ldA + half * 8;
  const bhalf* wP = W + (size_t)(ng * 64 + l15) * ldW + half * 8;
  const size_t w16 = (size_t)16 * ldW;

  v8f acc0 = {}, acc1 = {}, acc2 = {}, acc3 = {};
  for (int k0 = 0; k0 < K; k0 += 32) {
    v16bf a  = load_fragA(aP + k0);
    v16bf b0 = load_fragB(wP + k0);
    v16bf b1 = load_fragB(wP + k0 + w16);
    v16bf b2 = load_fragB(wP + k0 + 2 * w16);
    v16bf b3 = load_fragB(wP + k0 + 3 * w16);
    acc0 = __builtin_amdgcn_wmma_f32_16x16x32_bf16(false, a, false, b0, (short)0, acc0, false, false);
    acc1 = __builtin_amdgcn_wmma_f32_16x16x32_bf16(false, a, false, b1, (short)0, acc1, false, false);
    acc2 = __builtin_amdgcn_wmma_f32_16x16x32_bf16(false, a, false, b2, (short)0, acc2, false, false);
    acc3 = __builtin_amdgcn_wmma_f32_16x16x32_bf16(false, a, false, b3, (short)0, acc3, false, false);
  }
  float* o = out + (size_t)(tm * 16 + half * 8) * ldO + (ng * 64 + l15);
#pragma unroll
  for (int r = 0; r < 8; ++r) {
    o[(size_t)r * ldO +  0] = acc0[r];
    o[(size_t)r * ldO + 16] = acc1[r];
    o[(size_t)r * ldO + 32] = acc2[r];
    o[(size_t)r * ldO + 48] = acc3[r];
  }
}

// ---------------------------------------------------------------------------
// Logits GEMM: Hseq[1024,1024] (row m = t*16+b) * W_out[V,H]^T + b_out
// -> d_out[b, t, v] (f32). Same WMMA body, remapped store with bias.
// ---------------------------------------------------------------------------
__global__ void k_logits_gemm(const bhalf* __restrict__ Hseq,
                              const bhalf* __restrict__ Wout,
                              const float* __restrict__ bout,
                              float* __restrict__ out)
{
  const int lane = threadIdx.x & 31;
  const int wid  = (int)((blockIdx.x * blockDim.x + threadIdx.x) >> 5);
  const int nGrp = kV >> 6;                       // 500
  if (wid >= ((kB * kT) >> 4) * nGrp) return;     // 64 * 500 waves
  const int tm   = wid / nGrp;
  const int ng   = wid - tm * nGrp;
  const int half = lane >> 4;
  const int l15  = lane & 15;

  const bhalf* aP = Hseq + (size_t)(tm * 16 + l15) * kH + half * 8;
  const bhalf* wP = Wout + (size_t)(ng * 64 + l15) * kH + half * 8;
  const size_t w16 = (size_t)16 * kH;

  v8f acc0 = {}, acc1 = {}, acc2 = {}, acc3 = {};
  for (int k0 = 0; k0 < kH; k0 += 32) {
    v16bf a  = load_fragA(aP + k0);
    v16bf b0 = load_fragB(wP + k0);
    v16bf b1 = load_fragB(wP + k0 + w16);
    v16bf b2 = load_fragB(wP + k0 + 2 * w16);
    v16bf b3 = load_fragB(wP + k0 + 3 * w16);
    acc0 = __builtin_amdgcn_wmma_f32_16x16x32_bf16(false, a, false, b0, (short)0, acc0, false, false);
    acc1 = __builtin_amdgcn_wmma_f32_16x16x32_bf16(false, a, false, b1, (short)0, acc1, false, false);
    acc2 = __builtin_amdgcn_wmma_f32_16x16x32_bf16(false, a, false, b2, (short)0, acc2, false, false);
    acc3 = __builtin_amdgcn_wmma_f32_16x16x32_bf16(false, a, false, b3, (short)0, acc3, false, false);
  }
  const int n  = ng * 64 + l15;
  const float bi0 = bout[n], bi1 = bout[n + 16], bi2 = bout[n + 32], bi3 = bout[n + 48];
#pragma unroll
  for (int r = 0; r < 8; ++r) {
    const int m  = tm * 16 + half * 8 + r;        // m = t*16 + b
    const int bb = m & 15, tt = m >> 4;
    float* o = out + ((size_t)bb * kT + tt) * (size_t)kV + n;
    o[ 0] = acc0[r] + bi0;
    o[16] = acc1[r] + bi1;
    o[32] = acc2[r] + bi2;
    o[48] = acc3[r] + bi3;
  }
}

// ---------------------------------------------------------------------------
// Prologue helpers
// ---------------------------------------------------------------------------
__global__ void k_f32_to_bf16(const float* __restrict__ in, bhalf* __restrict__ out, long n)
{
  long i = (long)blockIdx.x * blockDim.x + threadIdx.x;
  long stride = (long)gridDim.x * blockDim.x;
  for (; i < n; i += stride) out[i] = (bhalf)in[i];
}

// in[R,C] f32 -> out[C,R] bf16
__global__ void k_transpose_bf16(const float* __restrict__ in, bhalf* __restrict__ out, int R, int C)
{
  long i = (long)blockIdx.x * blockDim.x + threadIdx.x;
  if (i >= (long)R * C) return;
  int r = (int)(i % R);
  int c = (int)(i / R);
  out[i] = (bhalf)in[(size_t)r * C + c];          // out[c*R + r]
}

// xemb[t][b][e] = bf16(emb[tgt[b,t], e])
__global__ void k_embed(const int* __restrict__ tgt, const float* __restrict__ emb,
                        bhalf* __restrict__ xemb)
{
  long i = (long)blockIdx.x * blockDim.x + threadIdx.x;
  if (i >= (long)kT * kB * kE) return;
  int e = (int)(i & (kE - 1));
  int b = (int)((i >> 9) & (kB - 1));
  int t = (int)(i >> 13);
  int tok = tgt[b * kT + t];
  xemb[i] = (bhalf)emb[(size_t)tok * kE + e];
}

__global__ void k_init_hidden(const float* __restrict__ hidden,
                              float* __restrict__ h0f, float* __restrict__ h1f,
                              bhalf* __restrict__ h0b, bhalf* __restrict__ h1b)
{
  int i = blockIdx.x * blockDim.x + threadIdx.x;
  if (i >= kB * kH) return;
  float a = hidden[i];
  float c = hidden[kB * kH + i];
  h0f[i] = a; h0b[i] = (bhalf)a;
  h1f[i] = c; h1b[i] = (bhalf)c;
}

// ---------------------------------------------------------------------------
// Bahdanau attention step (one block per batch row b):
//   scores[s] = sum_a v[a] * tanh(q[b,a] + kp[b,s,a]); softmax; ctx = attn @ enc
// Writes act0[b] = [ xemb_t[b] (bf16, 512) | bf16(ctx) (1024) ]
// ---------------------------------------------------------------------------
__global__ void k_attention(const float* __restrict__ q,     // [16,1024]
                            const float* __restrict__ kp,    // [16*64,1024]
                            const float* __restrict__ vatt,  // [1024]
                            const float* __restrict__ enc,   // [16,64,1024] f32
                            const bhalf* __restrict__ xemb_t,// [16,512] bf16
                            bhalf* __restrict__ act0)        // [16,1536] bf16
{
  __shared__ float s_sc[kS];
  __shared__ float s_at[kS];
  const int b = blockIdx.x;
  const int tid = threadIdx.x;               // 256 threads = 8 waves
  const int wv = tid >> 5, lane = tid & 31;

  for (int s = wv; s < kS; s += 8) {
    const float* kprow = kp + (size_t)(b * kS + s) * kA;
    const float* qrow  = q + (size_t)b * kA;
    float p = 0.f;
    for (int a = lane; a < kA; a += 32)
      p += vatt[a] * tanhf(qrow[a] + kprow[a]);
    for (int off = 16; off > 0; off >>= 1)
      p += __shfl_down(p, off, 32);
    if (lane == 0) s_sc[s] = p;
  }
  __syncthreads();
  if (tid == 0) {
    float mx = s_sc[0];
    for (int s = 1; s < kS; ++s) mx = fmaxf(mx, s_sc[s]);
    float sum = 0.f;
    for (int s = 0; s < kS; ++s) { float e = expf(s_sc[s] - mx); s_at[s] = e; sum += e; }
    float inv = 1.f / sum;
    for (int s = 0; s < kS; ++s) s_at[s] *= inv;
  }
  __syncthreads();
  for (int h = tid; h < kH; h += 256) {
    const float* e = enc + (size_t)b * kS * kH + h;
    float c = 0.f;
#pragma unroll 8
    for (int s = 0; s < kS; ++s) c += s_at[s] * e[(size_t)s * kH];
    act0[(size_t)b * kKin0 + kE + h] = (bhalf)c;
  }
  for (int e0 = tid; e0 < kE; e0 += 256)
    act0[(size_t)b * kKin0 + e0] = xemb_t[(size_t)b * kE + e0];
}

// ---------------------------------------------------------------------------
// GRU gate combine: r=sig(xr+hr), z=sig(xz+hz), n=tanh(xn + r*hn), h'=(1-z)n+z h
// Updates f32 master and bf16 GEMM copy; optionally records h' into Hseq.
// ---------------------------------------------------------------------------
__global__ void k_gru_combine(const float* __restrict__ gx, const float* __restrict__ gh,
                              const float* __restrict__ bih, const float* __restrict__ bhh,
                              float* __restrict__ hf, bhalf* __restrict__ hb,
                              bhalf* __restrict__ hseq)
{
  int i = blockIdx.x * blockDim.x + threadIdx.x;
  if (i >= kB * kH) return;
  const int bb = i >> 10, hh = i & (kH - 1);
  const float* gxr = gx + (size_t)bb * kG;
  const float* ghr = gh + (size_t)bb * kG;
  float xr = gxr[hh]          + bih[hh];
  float xz = gxr[kH + hh]     + bih[kH + hh];
  float xn = gxr[2 * kH + hh] + bih[2 * kH + hh];
  float hr = ghr[hh]          + bhh[hh];
  float hz = ghr[kH + hh]     + bhh[kH + hh];
  float hn = ghr[2 * kH + hh] + bhh[2 * kH + hh];
  float r = 1.f / (1.f + expf(-(xr + hr)));
  float z = 1.f / (1.f + expf(-(xz + hz)));
  float n = tanhf(xn + r * hn);
  float h = (1.f - z) * n + z * hf[i];
  hf[i] = h;
  hb[i] = (bhalf)h;
  if (hseq) hseq[i] = (bhalf)h;
}

// ---------------------------------------------------------------------------
// Host launcher
// ---------------------------------------------------------------------------
extern "C" void kernel_launch(void* const* d_in, const int* in_sizes, int n_in,
                              void* d_out, int out_size, void* d_ws, size_t ws_size,
                              hipStream_t stream)
{
  (void)in_sizes; (void)n_in; (void)out_size; (void)ws_size;

  const int*   tgt    = (const int*)  d_in[0];
  const float* hidden = (const float*)d_in[1];
  const float* enc    = (const float*)d_in[2];
  const float* emb    = (const float*)d_in[3];
  const float* Wq     = (const float*)d_in[4];
  const float* Wk     = (const float*)d_in[5];
  const float* vatt   = (const float*)d_in[6];
  const float* Wih0   = (const float*)d_in[7];
  const float* Whh0   = (const float*)d_in[8];
  const float* bih0   = (const float*)d_in[9];
  const float* bhh0   = (const float*)d_in[10];
  const float* Wih1   = (const float*)d_in[11];
  const float* Whh1   = (const float*)d_in[12];
  const float* bih1   = (const float*)d_in[13];
  const float* bhh1   = (const float*)d_in[14];
  const float* Wout   = (const float*)d_in[15];
  const float* bout   = (const float*)d_in[16];
  float* out = (float*)d_out;

  // ---- workspace carve-out (256B aligned blocks) ----
  char* ws = (char*)d_ws;
  size_t off = 0;
  auto wsAlloc = [&](size_t bytes) -> char* {
    char* p = ws + off;
    off += (bytes + 255) & ~(size_t)255;
    return p;
  };
  bhalf* wih0b = (bhalf*)wsAlloc((size_t)kG * kKin0 * 2);   //  9.4 MB
  bhalf* whh0b = (bhalf*)wsAlloc((size_t)kG * kH * 2);      //  6.3 MB
  bhalf* wih1b = (bhalf*)wsAlloc((size_t)kG * kH * 2);
  bhalf* whh1b = (bhalf*)wsAlloc((size_t)kG * kH * 2);
  bhalf* woutb = (bhalf*)wsAlloc((size_t)kV * kH * 2);      // 65.5 MB
  bhalf* wqTb  = (bhalf*)wsAlloc((size_t)kA * kH * 2);      //  2 MB  [A,H]
  bhalf* wkTb  = (bhalf*)wsAlloc((size_t)kA * kH * 2);      //  2 MB  [A,H]
  bhalf* encb  = (bhalf*)wsAlloc((size_t)kB * kS * kH * 2); //  2 MB
  float* kp    = (float*)wsAlloc((size_t)kB * kS * kA * 4); //  4 MB
  bhalf* xembb = (bhalf*)wsAlloc((size_t)kT * kB * kE * 2); //  1 MB
  bhalf* act0  = (bhalf*)wsAlloc((size_t)kB * kKin0 * 2);
  float* qbuf  = (float*)wsAlloc((size_t)kB * kA * 4);
  float* gx0   = (float*)wsAlloc((size_t)kB * kG * 4);
  float* gh0   = (float*)wsAlloc((size_t)kB * kG * 4);
  float* gx1   = (float*)wsAlloc((size_t)kB * kG * 4);
  float* gh1   = (float*)wsAlloc((size_t)kB * kG * 4);
  float* h0f   = (float*)wsAlloc((size_t)kB * kH * 4);
  float* h1f   = (float*)wsAlloc((size_t)kB * kH * 4);
  bhalf* h0b   = (bhalf*)wsAlloc((size_t)kB * kH * 2);
  bhalf* h1b   = (bhalf*)wsAlloc((size_t)kB * kH * 2);
  bhalf* hseq  = (bhalf*)wsAlloc((size_t)kT * kB * kH * 2); //  2 MB

  auto cvt = [&](const float* src, bhalf* dst, long n) {
    int blocks = (int)((n + 1023) / 1024);
    k_f32_to_bf16<<<blocks, 256, 0, stream>>>(src, dst, n);
  };
  auto gemm = [&](const bhalf* A, int ldA, const bhalf* W, int ldW,
                  float* O, int ldO, int M, int N, int K) {
    int waves  = (M / 16) * (N / 64);
    int blocks = (waves + 7) / 8;
    k_gemm_bf16<<<blocks, 256, 0, stream>>>(A, ldA, W, ldW, O, ldO, M, N, K);
  };

  // ---- prologue: weight conversion / transposes / embeddings / keys-proj ----
  cvt(Wih0, wih0b, (long)kG * kKin0);
  cvt(Whh0, whh0b, (long)kG * kH);
  cvt(Wih1, wih1b, (long)kG * kH);
  cvt(Whh1, whh1b, (long)kG * kH);
  cvt(Wout, woutb, (long)kV * kH);
  cvt(enc,  encb,  (long)kB * kS * kH);
  {
    long n = (long)kH * kA;
    int blocks = (int)((n + 255) / 256);
    k_transpose_bf16<<<blocks, 256, 0, stream>>>(Wq, wqTb, kH, kA);
    k_transpose_bf16<<<blocks, 256, 0, stream>>>(Wk, wkTb, kH, kA);
  }
  {
    long n = (long)kT * kB * kE;
    k_embed<<<(int)((n + 255) / 256), 256, 0, stream>>>(tgt, emb, xembb);
  }
  k_init_hidden<<<(kB * kH + 255) / 256, 256, 0, stream>>>(hidden, h0f, h1f, h0b, h1b);

  // keys_proj[b*S+s, a] = enc[b,s,:] @ Wk  (M=1024, N=1024, K=1024)
  gemm(encb, kH, wkTb, kH, kp, kA, kB * kS, kA, kH);

  // ---- sequential recurrence over T=64 steps ----
  for (int t = 0; t < kT; ++t) {
    // q = h1 @ Wq  (M=16, N=1024, K=1024)
    gemm(h1b, kH, wqTb, kH, qbuf, kA, kB, kA, kH);
    // attention + build act0 = [x_t | ctx]
    k_attention<<<kB, 256, 0, stream>>>(qbuf, kp, vatt, enc,
                                        xembb + (size_t)t * kB * kE, act0);
    // GRU layer 0 gates
    gemm(act0, kKin0, wih0b, kKin0, gx0, kG, kB, kG, kKin0);
    gemm(h0b,  kH,    whh0b, kH,    gh0, kG, kB, kG, kH);
    k_gru_combine<<<(kB * kH + 255) / 256, 256, 0, stream>>>(
        gx0, gh0, bih0, bhh0, h0f, h0b, (bhalf*)nullptr);
    // GRU layer 1 gates
    gemm(h0b, kH, wih1b, kH, gx1, kG, kB, kG, kH);
    gemm(h1b, kH, whh1b, kH, gh1, kG, kB, kG, kH);
    k_gru_combine<<<(kB * kH + 255) / 256, 256, 0, stream>>>(
        gx1, gh1, bih1, bhh1, h1f, h1b, hseq + (size_t)t * kB * kH);
  }

  // ---- epilogue: one big logits GEMM [1024,1024] x [1024,32000] + bias ----
  {
    int waves  = ((kB * kT) / 16) * (kV / 64);   // 64 * 500 = 32000
    int blocks = (waves + 7) / 8;                // 4000 blocks of 256
    k_logits_gemm<<<blocks, 256, 0, stream>>>(hseq, woutb, bout, out);
  }
}
